// DSBINModel_81011673137220
// MI455X (gfx1250) — matrix operand order repaired and checked by reading, not compiled, not run on hardware
//
#include <hip/hip_runtime.h>
#include <hip/hip_bf16.h>

// ---------------- problem constants (from reference) ----------------
#define Dh    52      // node hidden dim
#define D64   64      // padded node dim for WMMA K
#define EINh  14      // edge feature dim
#define K32   32      // padded edge-hidden dim for WMMA K
#define DDh   2704    // D*D
#define NLh   800     // ligand nodes
#define ELh   3200    // ligand edges
#define NPRh  6000    // protein nodes
#define EPh   24000   // protein edges
#define NBh   10      // distance bins

typedef _Float16 h16;
typedef __attribute__((ext_vector_type(16))) _Float16 v16h;
typedef __attribute__((ext_vector_type(2)))  _Float16 h2v;
typedef __attribute__((ext_vector_type(8)))  float    v8f;

__device__ __forceinline__ float sigmoidf_(float x){ return 1.0f/(1.0f+expf(-x)); }

// ====== edge MLP layer 1: hid32[E,32] = relu(efeat@eW1+eb1) zero-padded to K=32 ======
__global__ void edge_hidden_kernel(const float* __restrict__ efeat, const float* __restrict__ eW1,
                                   const float* __restrict__ eb1, h16* __restrict__ hid32, int E){
  int t = blockIdx.x*blockDim.x + threadIdx.x;
  if (t >= E*K32) return;
  int e = t >> 5, j = t & 31;
  float acc = 0.0f;
  if (j < EINh){
    acc = eb1[j];
    #pragma unroll
    for (int i = 0; i < EINh; ++i) acc += efeat[e*EINh + i] * eW1[i*EINh + j];
    acc = fmaxf(acc, 0.0f);
  }
  hid32[t] = (h16)acc;            // cols 14..31 stay exactly zero
}

// ====== eW2 -> f16, transposed + zero-padded: ew2t[2704][32] =========================
__global__ void ew2_pad_kernel(const float* __restrict__ eW2, h16* __restrict__ ew2t){
  int t = blockIdx.x*blockDim.x + threadIdx.x;
  if (t >= DDh*K32) return;
  int c = t >> 5, k = t & 31;
  ew2t[t] = (k < EINh) ? (h16)eW2[(size_t)k*DDh + c] : (h16)0.0f;
}

// ====== edge MLP layer 2 via WMMA: W[E,2704] = hid@eW2 + eb2 =========================
// One wave per 16x16 tile; branch-free packed fragment loads (data pre-padded).
__global__ void edge_w_wmma_kernel(const h16* __restrict__ hid32, const h16* __restrict__ ew2t,
                                   const float* __restrict__ eb2, h16* __restrict__ W){
  int lane = threadIdx.x;
  int half = lane >> 4, l16 = lane & 15;
  int e0  = blockIdx.x * 16;
  int col = blockIdx.y * 16 + l16;

  const h16* arow = hid32 + (size_t)(e0 + l16) * K32;  // A row (M = l16)
  const h16* bcol = ew2t  + (size_t)col * K32;         // B column (N = l16)
  v16h a, b;
  #pragma unroll
  for (int j = 0; j < 8; ++j){
    // A (16x32 f16): lanes0-15 K=0..15 ; lanes16-31 K=16..31
    int K0 = 16*half + 2*j;
    h2v va = *(const h2v*)(arow + K0);
    a[2*j] = va[0]; a[2*j+1] = va[1];
    // B (32x16 f16): VGPR j holds K = 16*half + 2j, 2j+1 ; N = lane&15
    int K1 = 16*half + 2*j;
    h2v vb = *(const h2v*)(bcol + K1);
    b[2*j] = vb[0]; b[2*j+1] = vb[1];
  }
  v8f c = {};
  c = __builtin_amdgcn_wmma_f32_16x16x32_f16(false, a, false, b, (short)0, c, false, false);
  float bias = eb2[col];
  #pragma unroll
  for (int r = 0; r < 8; ++r){
    int e = e0 + r + 8*half;                 // D layout: VGPR r -> M = r + 8*half
    W[(size_t)e*DDh + col] = (h16)(c[r] + bias);
  }
}

// ====== node init: out = relu(nfeat@W0+b0) ==========================================
__global__ void node_init_kernel(const float* __restrict__ nfeat, const float* __restrict__ W0,
                                 const float* __restrict__ b0, float* __restrict__ out, int N){
  int t = blockIdx.x*blockDim.x + threadIdx.x;
  if (t >= N*Dh) return;
  int n = t / Dh, d = t % Dh;
  float acc = b0[d];
  #pragma unroll 4
  for (int i = 0; i < Dh; ++i) acc += nfeat[n*Dh + i] * W0[i*Dh + d];
  out[t] = fmaxf(acc, 0.0f);
}

__global__ void zero_kernel(float* __restrict__ p, int n){
  int t = blockIdx.x*blockDim.x + threadIdx.x;
  if (t < n) p[t] = 0.0f;
}

// ====== per-edge matvec + scatter-sum into agg ======================================
__global__ void msg_kernel(const float* __restrict__ out, const h16* __restrict__ W,
                           const int* __restrict__ src, const int* __restrict__ dst,
                           float* __restrict__ agg){
  int e = blockIdx.x;
  int o = threadIdx.x;
  if (o >= Dh) return;
  int s = src[e], d = dst[e];
  const float* x  = out + (size_t)s*Dh;
  const h16*  We  = W   + (size_t)e*DDh;
  float acc = 0.f;
  #pragma unroll 4
  for (int i = 0; i < Dh; ++i) acc += x[i] * (float)We[i*Dh + o];   // coalesced over o
  atomicAdd(&agg[(size_t)d*Dh + o], acc);
}

// ====== m=relu(agg+out+bias); out' = [m,out]@Wm + bm ================================
__global__ void combine_kernel(const float* __restrict__ out, const float* __restrict__ agg,
                               const float* __restrict__ bias, const float* __restrict__ Wm,
                               const float* __restrict__ bm, float* __restrict__ outn, int N){
  int t = blockIdx.x*blockDim.x + threadIdx.x;
  if (t >= N*Dh) return;
  int n = t / Dh, o = t % Dh;
  const float* on = out + (size_t)n*Dh;
  const float* an = agg + (size_t)n*Dh;
  float acc = bm[o];
  for (int j = 0; j < Dh; ++j){
    float m = fmaxf(an[j] + on[j] + bias[j], 0.f);
    acc += m * Wm[j*Dh + o] + on[j] * Wm[(Dh + j)*Dh + o];
  }
  outn[t] = acc;
}

// ====== residual + stride-64 zero-padded f16 copy for the interaction GEMM ==========
__global__ void finalize_kernel(const float* __restrict__ out, const float* __restrict__ nfeat,
                                float* __restrict__ f, h16* __restrict__ f16, int N){
  int t = blockIdx.x*blockDim.x + threadIdx.x;
  if (t >= N*D64) return;
  int n = t >> 6, d = t & 63;
  float v = 0.0f;
  if (d < Dh){
    v = out[n*Dh + d] + nfeat[n*Dh + d];
    f[n*Dh + d] = v;
  }
  f16[t] = (h16)v;                 // cols 52..63 stay exactly zero
}

// ====== interaction map via WMMA: lig_f @ prot_f^T ; K=52 padded to 64 ==============
__global__ void imap_wmma_kernel(const h16* __restrict__ A16, const h16* __restrict__ B16,
                                 const float* __restrict__ ligLen, const float* __restrict__ protLen,
                                 float* __restrict__ ret_map, float* __restrict__ imap_t){
  int lane = threadIdx.x; int half = lane >> 4; int l16 = lane & 15;
  int n0 = blockIdx.x*16, p0 = blockIdx.y*16;
  const h16* arow = A16 + (size_t)(n0 + l16) * D64;
  const h16* brow = B16 + (size_t)(p0 + l16) * D64;
  v8f c = {};
  #pragma unroll
  for (int kk = 0; kk < 2; ++kk){
    v16h a, b;
    #pragma unroll
    for (int j = 0; j < 8; ++j){
      int K0 = kk*32 + 16*half + 2*j;
      h2v va = *(const h2v*)(arow + K0);
      a[2*j] = va[0]; a[2*j+1] = va[1];
      int K1 = kk*32 + 16*half + 2*j;
      h2v vb = *(const h2v*)(brow + K1);
      b[2*j] = vb[0]; b[2*j+1] = vb[1];
    }
    c = __builtin_amdgcn_wmma_f32_16x16x32_f16(false, a, false, b, (short)0, c, false, false);
  }
  int p = p0 + l16;
  float lp = protLen[p];
  #pragma unroll
  for (int r = 0; r < 8; ++r){
    int n = n0 + r + 8*half;
    float len = ligLen[n] * lp;
    float v = c[r];
    size_t idx = (size_t)n*NPRh + p;
    ret_map[idx] = len * v;
    imap_t[idx]  = len * tanhf(v);
  }
}

// ====== distance bin index per (n,p): 0..9 or 255 ===================================
__global__ void bins_kernel(const float* __restrict__ cl, const float* __restrict__ cp,
                            unsigned char* __restrict__ bins){
  int t = blockIdx.x*blockDim.x + threadIdx.x;
  if (t >= NLh*NPRh) return;
  int n = t / NPRh, p = t % NPRh;
  float dx = cl[n*3+0]-cp[p*3+0];
  float dy = cl[n*3+1]-cp[p*3+1];
  float dz = cl[n*3+2]-cp[p*3+2];
  float dist = sqrtf(dx*dx + dy*dy + dz*dz);
  unsigned char b = 255;
  if (dist > 0.f && dist <= 10.f){
    int bi2 = (int)ceilf(dist) - 1;            // bin k selects (k, k+1], 0-indexed
    bi2 = bi2 < 0 ? 0 : (bi2 > 9 ? 9 : bi2);
    b = (unsigned char)bi2;
  }
  bins[t] = b;
}

// ====== prot_primes[k,p,d] = wb[k] * sum_n [bin==k] imap_t[n,p]*lig_f[n,d] ==========
// Bins partition (n,p) -> 10 register accumulators per (p,d) thread; 832 thr/block.
__global__ void prot_primes_kernel(const float* __restrict__ imap_t, const unsigned char* __restrict__ bins,
                                   const float* __restrict__ ligf, const float* __restrict__ wb,
                                   float* __restrict__ pp){
  int t = threadIdx.x;                 // 16*52 = 832
  int pl = t & 15, d2 = t >> 4;
  int p = blockIdx.x*16 + pl;
  float acc[NBh];
  #pragma unroll
  for (int k = 0; k < NBh; ++k) acc[k] = 0.f;
  for (int n = 0; n < NLh; ++n){
    if ((n & 63) == 0 && n + 64 < NLh)
      __builtin_prefetch(&imap_t[(size_t)(n + 64)*NPRh + p], 0, 1);   // global_prefetch_b8
    float v = imap_t[(size_t)n*NPRh + p];
    int bk = bins[(size_t)n*NPRh + p];
    float contrib = v * ligf[n*Dh + d2];
    #pragma unroll
    for (int k = 0; k < NBh; ++k) acc[k] += (bk == k) ? contrib : 0.f;
  }
  #pragma unroll
  for (int k = 0; k < NBh; ++k) pp[((size_t)k*NPRh + p)*Dh + d2] = wb[k]*acc[k];
}

// ====== lig_primes[k,n,d] = wb[k] * sum_p [bin==k] imap_t[n,p]*prot_f[p,d] ==========
__global__ void lig_primes_kernel(const float* __restrict__ imap_t, const unsigned char* __restrict__ bins,
                                  const float* __restrict__ protf, const float* __restrict__ wb,
                                  float* __restrict__ lp){
  int t = threadIdx.x;
  int nl = t & 15, d2 = t >> 4;
  int n = blockIdx.x*16 + nl;
  float acc[NBh];
  #pragma unroll
  for (int k = 0; k < NBh; ++k) acc[k] = 0.f;
  const float* row = imap_t + (size_t)n*NPRh;
  const unsigned char* brow = bins + (size_t)n*NPRh;
  for (int p = 0; p < NPRh; ++p){
    if ((p & 127) == 0 && p + 128 < NPRh)
      __builtin_prefetch(&row[p + 128], 0, 1);                        // global_prefetch_b8
    float v = row[p];
    int bk = brow[p];
    float contrib = v * protf[p*Dh + d2];
    #pragma unroll
    for (int k = 0; k < NBh; ++k) acc[k] += (bk == k) ? contrib : 0.f;
  }
  #pragma unroll
  for (int k = 0; k < NBh; ++k) lp[((size_t)k*NLh + n)*Dh + d2] = wb[k]*acc[k];
}

// ====== wb = sigmoid(relu(1@fw1W+fw1b)@fw2W+fw2b) ===================================
__global__ void wb_kernel(const float* __restrict__ fw1W, const float* __restrict__ fw1b,
                          const float* __restrict__ fw2W, const float* __restrict__ fw2b,
                          float* __restrict__ wb){
  __shared__ float sw[NBh];
  int t = threadIdx.x;
  if (t < NBh){
    float a = fw1b[t];
    #pragma unroll
    for (int i = 0; i < NBh; ++i) a += fw1W[i*NBh + t];
    sw[t] = fmaxf(a, 0.f);
  }
  __syncthreads();
  if (t < NBh){
    float a = fw2b[t];
    #pragma unroll
    for (int i = 0; i < NBh; ++i) a += sw[i]*fw2W[i*NBh + t];
    wb[t] = sigmoidf_(a);
  }
}

// ====== Set2Set pooling: 1 block per (bin, side) ====================================
__global__ void set2set_kernel(const float* __restrict__ lig_primes, const float* __restrict__ prot_primes,
                               const float* __restrict__ Wh_l, const float* __restrict__ Wi_l,
                               const float* __restrict__ bh_l, const float* __restrict__ bi_l,
                               const float* __restrict__ Wh_p, const float* __restrict__ Wi_p,
                               const float* __restrict__ bh_p, const float* __restrict__ bi_p,
                               float* __restrict__ lig_emb, float* __restrict__ prot_emb){
  int k = blockIdx.x, side = blockIdx.y;
  int nset = side ? NPRh : NLh;
  const float* x  = side ? (prot_primes + (size_t)k*NPRh*Dh) : (lig_primes + (size_t)k*NLh*Dh);
  const float* Wi = side ? Wi_p : Wi_l;
  const float* Wh = side ? Wh_p : Wh_l;
  const float* bi = side ? bi_p : bi_l;
  const float* bh = side ? bh_p : bh_l;
  float* emb = (side ? prot_emb : lig_emb) + (size_t)k*104;

  __shared__ float sl[NPRh];
  __shared__ float q[104], hs[Dh], cs[Dh], g[208], red[256];
  int tid = threadIdx.x;
  for (int i = tid; i < 104; i += 256) q[i] = 0.f;
  if (tid < Dh){ hs[tid] = 0.f; cs[tid] = 0.f; }
  __syncthreads();

  for (int it = 0; it < 2; ++it){
    // LSTM gates: g = q@Wi + bi + h@Wh + bh
    if (tid < 208){
      float a = bi[tid] + bh[tid];
      for (int j = 0; j < 104; ++j) a += q[j]  * Wi[j*208 + tid];
      for (int j = 0; j < Dh;  ++j) a += hs[j] * Wh[j*208 + tid];
      g[tid] = a;
    }
    __syncthreads();
    if (tid < Dh){
      float ig = sigmoidf_(g[tid]);
      float fg = sigmoidf_(g[Dh + tid]);
      float gc = tanhf   (g[2*Dh + tid]);
      float og = sigmoidf_(g[3*Dh + tid]);
      float cn = fg*cs[tid] + ig*gc;
      cs[tid] = cn;
      hs[tid] = og*tanhf(cn);
    }
    __syncthreads();
    // logits e_n = x[n].h
    for (int n = tid; n < nset; n += 256){
      const float* xr = x + (size_t)n*Dh;
      float a = 0.f;
      #pragma unroll 4
      for (int j = 0; j < Dh; ++j) a += xr[j]*hs[j];
      sl[n] = a;
    }
    __syncthreads();
    // max
    float lm = -3.402823e38f;
    for (int n = tid; n < nset; n += 256) lm = fmaxf(lm, sl[n]);
    red[tid] = lm; __syncthreads();
    for (int s = 128; s > 0; s >>= 1){ if (tid < s) red[tid] = fmaxf(red[tid], red[tid+s]); __syncthreads(); }
    float mx = red[0]; __syncthreads();
    // exp + sum
    float ls = 0.f;
    for (int n = tid; n < nset; n += 256){ float e = expf(sl[n]-mx); sl[n] = e; ls += e; }
    red[tid] = ls; __syncthreads();
    for (int s = 128; s > 0; s >>= 1){ if (tid < s) red[tid] += red[tid+s]; __syncthreads(); }
    float Z = red[0]; __syncthreads();
    // r[d] = (1/Z) sum_n exp_n * x[n,d]  (4 partial groups x 52 dims)
    if (tid < 208){
      int dd2 = tid % Dh, grp = tid / Dh;
      float a = 0.f;
      for (int n = grp; n < nset; n += 4) a += sl[n]*x[(size_t)n*Dh + dd2];
      g[tid] = a;
    }
    __syncthreads();
    if (tid < Dh){
      float r = (g[tid] + g[Dh+tid] + g[2*Dh+tid] + g[3*Dh+tid]) / Z;
      q[tid]      = hs[tid];
      q[Dh + tid] = r;
    }
    __syncthreads();
  }
  for (int i = tid; i < 104; i += 256) emb[i] = q[i];
}

// ====== head MLP: fc1-relu, fc2, bn, relu, fc3 ======================================
__global__ void head_kernel(const float* __restrict__ lig_emb, const float* __restrict__ prot_emb,
                            const float* __restrict__ fc1W, const float* __restrict__ fc1b,
                            const float* __restrict__ fc2W, const float* __restrict__ fc2b,
                            const float* __restrict__ fc3W, const float* __restrict__ fc3b,
                            const float* __restrict__ bn_g, const float* __restrict__ bn_b,
                            const float* __restrict__ bn_m, const float* __restrict__ bn_v,
                            float* __restrict__ pred){
  __shared__ float fin[2080], x1[256], x2[64];
  int tid = threadIdx.x;
  for (int t = tid; t < 2080; t += 256){
    int k = t / 208, j = t % 208;
    fin[t] = (j < 104) ? lig_emb[k*104 + j] : prot_emb[k*104 + (j-104)];
  }
  __syncthreads();
  {
    float a = fc1b[tid];
    for (int j = 0; j < 2080; ++j) a += fin[j]*fc1W[j*256 + tid];
    x1[tid] = fmaxf(a, 0.f);
  }
  __syncthreads();
  if (tid < 64){
    float a = fc2b[tid];
    for (int j = 0; j < 256; ++j) a += x1[j]*fc2W[j*64 + tid];
    a = bn_g[tid]*(a - bn_m[tid])*rsqrtf(bn_v[tid] + 1e-5f) + bn_b[tid];
    x2[tid] = fmaxf(a, 0.f);
  }
  __syncthreads();
  if (tid == 0){
    float a = fc3b[0];
    for (int j = 0; j < 64; ++j) a += x2[j]*fc3W[j];
    pred[0] = a;
  }
}

// ================================ host side =========================================
static void run_gather(const float* nfeat, const float* efeat, const int* src, const int* dst,
                       const float* W0, const float* Wm, const float* b0, const float* bias,
                       const float* bm, const float* eW1, const float* eW2, const float* eb1,
                       const float* eb2, int N, int E,
                       h16* hid32, h16* ew2t, h16* W, float* outA, float* outB, float* agg,
                       float* f_out, h16* f16_out, hipStream_t stream)
{
  int t1 = E*K32;
  edge_hidden_kernel<<<(t1+255)/256, 256, 0, stream>>>(efeat, eW1, eb1, hid32, E);
  int t2 = DDh*K32;
  ew2_pad_kernel<<<(t2+255)/256, 256, 0, stream>>>(eW2, ew2t);
  dim3 g2(E/16, DDh/16);
  edge_w_wmma_kernel<<<g2, 32, 0, stream>>>(hid32, ew2t, eb2, W);
  int t3 = N*Dh;
  node_init_kernel<<<(t3+255)/256, 256, 0, stream>>>(nfeat, W0, b0, outA, N);
  float* cur = outA; float* nxt = outB;
  for (int s = 0; s < 4; ++s){
    zero_kernel<<<(t3+255)/256, 256, 0, stream>>>(agg, t3);
    msg_kernel<<<E, 64, 0, stream>>>(cur, W, src, dst, agg);
    combine_kernel<<<(t3+255)/256, 256, 0, stream>>>(cur, agg, bias, Wm, bm, nxt, N);
    float* tmp = cur; cur = nxt; nxt = tmp;
  }
  int t4 = N*D64;
  finalize_kernel<<<(t4+255)/256, 256, 0, stream>>>(cur, nfeat, f_out, f16_out, N);
}

extern "C" void kernel_launch(void* const* d_in, const int* in_sizes, int n_in,
                              void* d_out, int out_size, void* d_ws, size_t ws_size,
                              hipStream_t stream) {
  (void)in_sizes; (void)n_in; (void)out_size; (void)ws_size;
  // Top-level dict in insertion order; nested param dicts in sorted-key (pytree) order.
  const float* nfeat_l = (const float*)d_in[0];
  const float* efeat_l = (const float*)d_in[1];
  const int*   src_l   = (const int*)  d_in[2];
  const int*   dst_l   = (const int*)  d_in[3];
  const float* coord_l = (const float*)d_in[4];
  const float* lig_len = (const float*)d_in[5];
  const float* nfeat_p = (const float*)d_in[6];
  const float* efeat_p = (const float*)d_in[7];
  const int*   src_p   = (const int*)  d_in[8];
  const int*   dst_p   = (const int*)  d_in[9];
  const float* coord_p = (const float*)d_in[10];
  const float* prot_len= (const float*)d_in[11];
  // gp_l sorted keys: W0, Wm, b0, bias, bm, eW1, eW2, eb1, eb2
  const float* W0_l  =(const float*)d_in[12]; const float* Wm_l  =(const float*)d_in[13];
  const float* b0_l  =(const float*)d_in[14]; const float* bias_l=(const float*)d_in[15];
  const float* bm_l  =(const float*)d_in[16]; const float* eW1_l =(const float*)d_in[17];
  const float* eW2_l =(const float*)d_in[18]; const float* eb1_l =(const float*)d_in[19];
  const float* eb2_l =(const float*)d_in[20];
  const float* W0_p  =(const float*)d_in[21]; const float* Wm_p  =(const float*)d_in[22];
  const float* b0_p  =(const float*)d_in[23]; const float* bias_p=(const float*)d_in[24];
  const float* bm_p  =(const float*)d_in[25]; const float* eW1_p =(const float*)d_in[26];
  const float* eW2_p =(const float*)d_in[27]; const float* eb1_p =(const float*)d_in[28];
  const float* eb2_p =(const float*)d_in[29];
  // s2s sorted keys: Wh, Wi, bh, bi
  const float* Wh_l=(const float*)d_in[30]; const float* Wi_l=(const float*)d_in[31];
  const float* bh_l=(const float*)d_in[32]; const float* bi_l=(const float*)d_in[33];
  const float* Wh_p=(const float*)d_in[34]; const float* Wi_p=(const float*)d_in[35];
  const float* bh_p=(const float*)d_in[36]; const float* bi_p=(const float*)d_in[37];
  // head sorted keys: bn_b,bn_g,bn_m,bn_v,fc1W,fc1b,fc2W,fc2b,fc3W,fc3b,fw1W,fw1b,fw2W,fw2b
  const float* bn_b=(const float*)d_in[38]; const float* bn_g=(const float*)d_in[39];
  const float* bn_m=(const float*)d_in[40]; const float* bn_v=(const float*)d_in[41];
  const float* fc1W=(const float*)d_in[42]; const float* fc1b=(const float*)d_in[43];
  const float* fc2W=(const float*)d_in[44]; const float* fc2b=(const float*)d_in[45];
  const float* fc3W=(const float*)d_in[46]; const float* fc3b=(const float*)d_in[47];
  const float* fw1W=(const float*)d_in[48]; const float* fw1b=(const float*)d_in[49];
  const float* fw2W=(const float*)d_in[50]; const float* fw2b=(const float*)d_in[51];

  // ---------------- workspace carve ----------------
  char* base = (char*)d_ws;
  size_t off = 0;
  auto carve = [&](size_t bytes)->char*{
    char* p = base + off;
    off = (off + bytes + 255) & ~(size_t)255;
    return p;
  };
  h16*   hid_l   = (h16*)  carve((size_t)ELh*K32*sizeof(h16));
  h16*   hid_p   = (h16*)  carve((size_t)EPh*K32*sizeof(h16));
  h16*   ew2t_l  = (h16*)  carve((size_t)DDh*K32*sizeof(h16));
  h16*   ew2t_p  = (h16*)  carve((size_t)DDh*K32*sizeof(h16));
  h16*   W_l     = (h16*)  carve((size_t)ELh*DDh*sizeof(h16));
  h16*   W_p     = (h16*)  carve((size_t)EPh*DDh*sizeof(h16));
  float* outA_l  = (float*)carve((size_t)NLh*Dh*sizeof(float));
  float* outB_l  = (float*)carve((size_t)NLh*Dh*sizeof(float));
  float* agg_l   = (float*)carve((size_t)NLh*Dh*sizeof(float));
  float* outA_p  = (float*)carve((size_t)NPRh*Dh*sizeof(float));
  float* outB_p  = (float*)carve((size_t)NPRh*Dh*sizeof(float));
  float* agg_p   = (float*)carve((size_t)NPRh*Dh*sizeof(float));
  float* ligf    = (float*)carve((size_t)NLh*Dh*sizeof(float));
  float* protf   = (float*)carve((size_t)NPRh*Dh*sizeof(float));
  h16*   ligf16  = (h16*)  carve((size_t)NLh*D64*sizeof(h16));
  h16*   protf16 = (h16*)  carve((size_t)NPRh*D64*sizeof(h16));
  float* imap_t  = (float*)carve((size_t)NLh*NPRh*sizeof(float));
  unsigned char* binsp = (unsigned char*)carve((size_t)NLh*NPRh);
  float* wbp     = (float*)carve(NBh*sizeof(float));
  float* lig_pr  = (float*)carve((size_t)NBh*NLh*Dh*sizeof(float));
  float* prot_pr = (float*)carve((size_t)NBh*NPRh*Dh*sizeof(float));
  float* lig_emb = (float*)carve((size_t)NBh*104*sizeof(float));
  float* prot_emb= (float*)carve((size_t)NBh*104*sizeof(float));

  float* pred    = (float*)d_out;
  float* ret_map = pred + 1;

  // ---------------- GatherModels (NNConv x4, WMMA edge weights) ----------------
  run_gather(nfeat_l, efeat_l, src_l, dst_l, W0_l, Wm_l, b0_l, bias_l, bm_l,
             eW1_l, eW2_l, eb1_l, eb2_l, NLh, ELh,
             hid_l, ew2t_l, W_l, outA_l, outB_l, agg_l, ligf, ligf16, stream);
  run_gather(nfeat_p, efeat_p, src_p, dst_p, W0_p, Wm_p, b0_p, bias_p, bm_p,
             eW1_p, eW2_p, eb1_p, eb2_p, NPRh, EPh,
             hid_p, ew2t_p, W_p, outA_p, outB_p, agg_p, protf, protf16, stream);

  // ---------------- bin weights, interaction map (WMMA), distance bins --------
  wb_kernel<<<1, 32, 0, stream>>>(fw1W, fw1b, fw2W, fw2b, wbp);
  dim3 gi(NLh/16, NPRh/16);
  imap_wmma_kernel<<<gi, 32, 0, stream>>>(ligf16, protf16, lig_len, prot_len, ret_map, imap_t);
  int tb = NLh*NPRh;
  bins_kernel<<<(tb+255)/256, 256, 0, stream>>>(coord_l, coord_p, binsp);

  // ---------------- binned primes (fused masks, 10 register accumulators) -----
  prot_primes_kernel<<<NPRh/16, 16*Dh, 0, stream>>>(imap_t, binsp, ligf, wbp, prot_pr);
  lig_primes_kernel <<<NLh/16,  16*Dh, 0, stream>>>(imap_t, binsp, protf, wbp, lig_pr);

  // ---------------- Set2Set x (10 bins x 2 sides) + head MLP -------------------
  dim3 gs(NBh, 2);
  set2set_kernel<<<gs, 256, 0, stream>>>(lig_pr, prot_pr,
                                         Wh_l, Wi_l, bh_l, bi_l,
                                         Wh_p, Wi_p, bh_p, bi_p,
                                         lig_emb, prot_emb);
  head_kernel<<<1, 256, 0, stream>>>(lig_emb, prot_emb, fc1W, fc1b, fc2W, fc2b,
                                     fc3W, fc3b, bn_g, bn_b, bn_m, bn_v, pred);
}
